// BubblePredictor_44830868635915
// MI455X (gfx1250) — matrix operand order repaired and checked by reading, not compiled, not run on hardware
//
#include <hip/hip_runtime.h>
#include <hip/hip_bf16.h>
#include <math.h>

// ---------------------------------------------------------------------------
// LSTM (B=512, T=512, D=8, H=256) for MI455X / gfx1250 (wave32, WMMA).
//
// 32 WGs x 16 batch rows; each WG runs all 512 steps (batch rows are
// independent sequences). Per step: [16 x 288] x [288 x 1024] f16 GEMM with
// f32 accumulation (v_wmma_f32_16x16x32_f16), K = [h(256)|x(8)|pad] = 9
// k-tiles. 16 waves/WG; wave w owns N-tiles {w,16+w,32+w,48+w} = gates
// i,f,g,o for hidden units 16w..16w+15 -> cell update is lane-local, cell
// state c lives in 8 VGPRs/lane for the whole kernel.
//
// Weight residency:
//   - k-tiles 0..3: B-fragments pinned in registers (16 frags = 128 VGPRs).
//   - k-tiles 4..8: streamed from L2 each step as coalesced 32B/lane
//     global_load_b128 (saddr form), software-pipelined: first streamed
//     k-tile is fetched before the resident WMMA section, then kt+1 is
//     fetched while kt is consumed. A 64-bit offset (not the pointer) is
//     laundered through an empty asm each iteration so the loads are
//     re-issued instead of hoisted-and-spilled, while keeping the global
//     (addrspace 1) load path instead of flat.
// h is exchanged between waves through LDS in a contiguous-k A-fragment layout
// (lane half h holds k 16h..16h+15, as in the B pack) so A loads are 2x ds_load_b128.
// ---------------------------------------------------------------------------

typedef _Float16 half_t;
typedef __attribute__((ext_vector_type(16))) _Float16 v16h;
typedef __attribute__((ext_vector_type(8)))  float    v8f;

#define T_STEPS 512
#define HID     256
#define DIN     8
#define KTILES  9          // ceil((256+8)/32)
#define KRES    4          // k-tiles with register-resident B fragments
#define NTILES  64         // 4H / 16
#define WGB     16         // batch rows per workgroup
#define NTH     512        // 16 waves

// d_out layout (flat, return order): logits[B][T][2], h_T[B][H], c_T[B][H]
#define HOFF  (512 * 512 * 2)
#define COFF  (HOFF + 512 * 256)

// ---------------------------------------------------------------------------
// Prep: pack [W_hh | W_ih] (f32) -> f16 B-fragment layout in d_ws.
// Fragment (kt, nt), lane l, 16 halves: n = nt*16 + (l&15),
// k = kt*32 + (l>>4)*16 + hidx.  Column n maps directly to gate-major rows
// of the reference weights (i:0-255, f:256-511, g:512-767, o:768-1023).
// Wcat[n][k]: k<256 -> W_hh[n][k]; 256<=k<264 -> W_ih[n][k-256]; else 0.
// ---------------------------------------------------------------------------
__global__ void lstm_pack_weights(const float* __restrict__ W_ih,
                                  const float* __restrict__ W_hh,
                                  half_t* __restrict__ pack) {
    int gid = blockIdx.x * 256 + threadIdx.x;        // (kt*64+nt)*32 + l
    if (gid >= KTILES * NTILES * 32) return;
    int kt  = gid >> 11;
    int rem = gid & 2047;
    int l   = rem & 31;
    int n   = (rem >> 5) * 16 + (l & 15);
    int kb  = kt * 32 + (l >> 4) * 16;

    v16h frag;
#pragma unroll
    for (int hidx = 0; hidx < 16; ++hidx) {
        int k = kb + hidx;
        float v = 0.0f;
        if (k < HID)            v = W_hh[n * HID + k];
        else if (k < HID + DIN) v = W_ih[n * DIN + (k - HID)];
        frag[hidx] = (half_t)v;
    }
    ((v16h*)pack)[gid] = frag;
}

__device__ __forceinline__ float sigm(float x) {
    return 1.0f / (1.0f + __expf(-x));
}

#define WMMA(A, B, C) \
    __builtin_amdgcn_wmma_f32_16x16x32_f16(false, (A), false, (B), (short)0, (C), false, false)

// ---------------------------------------------------------------------------
// Persistent LSTM kernel: 32 blocks x 512 threads (16 wave32 / WGP).
// ---------------------------------------------------------------------------
__global__
__attribute__((amdgpu_flat_work_group_size(NTH, NTH), amdgpu_waves_per_eu(4)))
void lstm_kernel(const float*  __restrict__ history,   // [512][512][8]
                 const half_t* __restrict__ Wpack,     // packed B fragments
                 const float*  __restrict__ W_out,     // [2][256]
                 const float*  __restrict__ b_out,     // [2]
                 float*        __restrict__ out) {
    // A-fragment staging: [kt][lane][16 halves] (32B per lane per fragment)
    __shared__ __align__(32) half_t hbuf[KTILES * 32 * 16];   // 9216 B
    __shared__ float hplain[WGB * HID];                       // 16 KB (head)
    __shared__ float wout[2 * HID];                           // 2 KB

    const int tid = threadIdx.x;
    const int w   = tid >> 5;        // wave 0..15
    const int l   = tid & 31;
    const int li  = l & 15;
    const int hi  = l >> 4;
    const int wg  = blockIdx.x;

    // --- one-time init -----------------------------------------------------
    wout[tid] = W_out[tid];                           // 512 floats
    for (int i = tid; i < KTILES * 512; i += NTH) hbuf[i]   = (half_t)0.0f;
    for (int i = tid; i < WGB * HID;   i += NTH) hplain[i] = 0.0f;
    __syncthreads();
    if (tid < WGB * DIN) {                            // x_0 -> kt=8, kk=0..7
        int b = tid >> 3, d = tid & 7;
        hbuf[8 * 512 + b * 16 + d] =
            (half_t)history[(size_t)(wg * WGB + b) * (T_STEPS * DIN) + d];
    }
    const float b0 = b_out[0], b1 = b_out[1];

    float c[8];
#pragma unroll
    for (int v = 0; v < 8; ++v) c[v] = 0.0f;

    // lane-constant addressing for the h -> A-fragment scatter
    // element (b=M, j): kt=j>>5, kk=j&31; lanes 0-15 hold K{0-15},
    // lanes 16-31 hold K{16-31}; hidx = kk&15
    const int j      = 16 * w + li;                   // this lane's hidden col
    const int kt_h   = w >> 1;
    const int la_off = (w & 1) * 16;
    const int hidx_h = li;
    const int bbase  = hi * 8;

    const v16h* Bp = (const v16h*)Wpack;
    const int nt0 = w, nt1 = 16 + w, nt2 = 32 + w, nt3 = 48 + w;

    // --- register-resident B fragments: k-tiles 0..KRES-1 (128 VGPRs) ------
    v16h breg[KRES * 4];
#pragma unroll
    for (int kt = 0; kt < KRES; ++kt) {
        breg[kt * 4 + 0] = Bp[(kt * NTILES + nt0) * 32 + l];
        breg[kt * 4 + 1] = Bp[(kt * NTILES + nt1) * 32 + l];
        breg[kt * 4 + 2] = Bp[(kt * NTILES + nt2) * 32 + l];
        breg[kt * 4 + 3] = Bp[(kt * NTILES + nt3) * 32 + l];
    }

    __syncthreads();

    for (int t = 0; t < T_STEPS; ++t) {
        // Launder a 64-bit offset (kept in SGPRs) so the streamed loads
        // below cannot be hoisted out of the t-loop, while Bp keeps its
        // global address-space inference (global_load_b128, saddr form).
        long off = 0;
        asm volatile("" : "+s"(off));
        const v16h* Bt = Bp + off;

        // ---- prefetch first streamed k-tile (in flight during resident
        //      WMMA section below) -------------------------------------
        v16h s0cur = Bt[(KRES * NTILES + nt0) * 32 + l];
        v16h s1cur = Bt[(KRES * NTILES + nt1) * 32 + l];
        v16h s2cur = Bt[(KRES * NTILES + nt2) * 32 + l];
        v16h s3cur = Bt[(KRES * NTILES + nt3) * 32 + l];

        // ---- GEMM: gates(t) = [h_{t-1} | x_t] @ Wcat ----------------------
        v8f a0 = {}, a1 = {}, a2 = {}, a3 = {};
#pragma unroll
        for (int kt = 0; kt < KRES; ++kt) {           // register-resident B
            v16h afrag = *(const v16h*)&hbuf[kt * 512 + l * 16];
            a0 = WMMA(afrag, breg[kt * 4 + 0], a0);
            a1 = WMMA(afrag, breg[kt * 4 + 1], a1);
            a2 = WMMA(afrag, breg[kt * 4 + 2], a2);
            a3 = WMMA(afrag, breg[kt * 4 + 3], a3);
        }
#pragma unroll
        for (int kt = KRES; kt < KTILES; ++kt) {      // streamed, pipelined
            if (kt + 1 < KTILES) {                    // fetch kt+1
                v16h s0n = Bt[((kt + 1) * NTILES + nt0) * 32 + l];
                v16h s1n = Bt[((kt + 1) * NTILES + nt1) * 32 + l];
                v16h s2n = Bt[((kt + 1) * NTILES + nt2) * 32 + l];
                v16h s3n = Bt[((kt + 1) * NTILES + nt3) * 32 + l];
                v16h afrag = *(const v16h*)&hbuf[kt * 512 + l * 16];
                a0 = WMMA(afrag, s0cur, a0);
                a1 = WMMA(afrag, s1cur, a1);
                a2 = WMMA(afrag, s2cur, a2);
                a3 = WMMA(afrag, s3cur, a3);
                s0cur = s0n; s1cur = s1n; s2cur = s2n; s3cur = s3n;
            } else {
                v16h afrag = *(const v16h*)&hbuf[kt * 512 + l * 16];
                a0 = WMMA(afrag, s0cur, a0);
                a1 = WMMA(afrag, s1cur, a1);
                a2 = WMMA(afrag, s2cur, a2);
                a3 = WMMA(afrag, s3cur, a3);
            }
        }

        // ---- head for step t-1 (reads hplain = h_{t-1}) -------------------
        if (t > 0) {
            float p0 = 0.0f, p1 = 0.0f;
#pragma unroll
            for (int it = 0; it < 8; ++it) {
                float hv = hplain[w * HID + l + it * 32];
                p0 += hv * wout[l + it * 32];
                p1 += hv * wout[HID + l + it * 32];
            }
            for (int off2 = 16; off2 > 0; off2 >>= 1) {
                p0 += __shfl_xor(p0, off2);
                p1 += __shfl_xor(p1, off2);
            }
            if (l == 0) {
                float* lg = out + (((size_t)(wg * WGB + w)) * T_STEPS + (t - 1)) * 2;
                lg[0] = p0 + b0;
                lg[1] = p1 + b1;
            }
        }
        __syncthreads();   // all reads of hbuf/hplain for step t done

        // ---- lane-local LSTM cell update: i,f,g,o live in a0..a3 ----------
#pragma unroll
        for (int v = 0; v < 8; ++v) {
            float iv = sigm(a0[v]);
            float fv = sigm(a1[v]);
            float gv = tanhf(a2[v]);
            float ov = sigm(a3[v]);
            c[v] = fv * c[v] + iv * gv;
            float hv = ov * tanhf(c[v]);
            int b = bbase + v;
            hplain[b * HID + j] = hv;
            hbuf[kt_h * 512 + (b + la_off) * 16 + hidx_h] = (half_t)hv;
        }
        // ---- stage x_{t+1} into kt=8 --------------------------------------
        if (t + 1 < T_STEPS && tid < WGB * DIN) {
            int b = tid >> 3, d = tid & 7;
            hbuf[8 * 512 + b * 16 + d] =
                (half_t)history[(size_t)(wg * WGB + b) * (T_STEPS * DIN)
                                + (t + 1) * DIN + d];
        }
        __syncthreads();   // h_t / x_{t+1} visible to all waves
    }

    // ---- head for final step (h_{T-1} in hplain) --------------------------
    {
        float p0 = 0.0f, p1 = 0.0f;
#pragma unroll
        for (int it = 0; it < 8; ++it) {
            float hv = hplain[w * HID + l + it * 32];
            p0 += hv * wout[l + it * 32];
            p1 += hv * wout[HID + l + it * 32];
        }
        for (int off2 = 16; off2 > 0; off2 >>= 1) {
            p0 += __shfl_xor(p0, off2);
            p1 += __shfl_xor(p1, off2);
        }
        if (l == 0) {
            float* lg = out + (((size_t)(wg * WGB + w)) * T_STEPS + (T_STEPS - 1)) * 2;
            lg[0] = p0 + b0;
            lg[1] = p1 + b1;
        }
    }

    // ---- h_T from hplain, c_T from registers ------------------------------
    for (int i = tid; i < WGB * HID; i += NTH) {
        int b = i >> 8, jj = i & 255;
        out[HOFF + (size_t)(wg * WGB + b) * HID + jj] = hplain[i];
    }
#pragma unroll
    for (int v = 0; v < 8; ++v) {
        int bg = wg * WGB + bbase + v;
        out[COFF + (size_t)bg * HID + j] = c[v];
    }
}

extern "C" void kernel_launch(void* const* d_in, const int* in_sizes, int n_in,
                              void* d_out, int out_size, void* d_ws, size_t ws_size,
                              hipStream_t stream) {
    (void)in_sizes; (void)n_in; (void)out_size; (void)ws_size;
    const float* history = (const float*)d_in[0];   // [512,512,8]
    const float* W_ih    = (const float*)d_in[1];   // [1024,8]
    const float* W_hh    = (const float*)d_in[2];   // [1024,256]
    const float* W_out   = (const float*)d_in[3];   // [2,256]
    const float* b_out   = (const float*)d_in[4];   // [2]
    half_t* pack = (half_t*)d_ws;                   // 576 KB packed f16 weights

    int pack_threads = KTILES * NTILES * 32;        // 18432
    lstm_pack_weights<<<(pack_threads + 255) / 256, 256, 0, stream>>>(W_ih, W_hh, pack);
    lstm_kernel<<<512 / WGB, NTH, 0, stream>>>(history, pack, W_out, b_out,
                                               (float*)d_out);
}
